// VNAttention_12343736009199
// MI455X (gfx1250) — matrix operand order repaired and checked by reading, not compiled, not run on hardware
//
#include <hip/hip_runtime.h>
#include <hip/hip_bf16.h>

#define BB 4
#define CC 256
#define NN 2048
#define HH 8
#define HD 32
#define K96 96

#define KPAD 104   // padded K-tile row (halves): 52 dwords, conflict-free over 16 rows
#define VPAD 40    // padded V-tile row (halves): 20 dwords, conflict-free over 16 rows

typedef __attribute__((ext_vector_type(16))) __bf16 v16bf;
typedef __attribute__((ext_vector_type(8)))  __bf16 v8bf;
typedef __attribute__((ext_vector_type(8)))  float  v8f;

static __device__ __forceinline__ v16bf combine16(uint4 a, uint4 b) {
    union { struct { uint4 a, b; } s; v16bf v; } u;
    u.s.a = a; u.s.b = b;
    return u.v;
}

// A-operand bf16 load: halves [p, p+8) and [p+16, p+24)   (K = 8*hi+e / 16+8*hi+(e-8))
static __device__ __forceinline__ v16bf ldA(const __bf16* p) {
    return combine16(*(const uint4*)p, *(const uint4*)(p + 16));
}
// B-operand bf16 load: halves [p, p+8) and [p+16, p+24)   (K = 8*hi+e / 16+8*hi+(e-8))
static __device__ __forceinline__ v16bf ldB(const __bf16* p) {
    return combine16(*(const uint4*)p, *(const uint4*)(p + 16));
}

#define WMMA_BF16(A, Bm, Cm) \
    __builtin_amdgcn_wmma_f32_16x16x32_bf16(false, (A), false, (Bm), (short)0, (Cm), false, false)

// ---------------- Kernel W: convert 4 weight matrices fp32 -> bf16 -------------
__global__ void vn_cvt_w(const float* __restrict__ Wq, const float* __restrict__ Wk,
                         const float* __restrict__ Wv, const float* __restrict__ Wo,
                         __bf16* __restrict__ Wb) {
    const int idx = blockIdx.x * blockDim.x + threadIdx.x;   // 0 .. 4*65536-1
    const int m = idx >> 16, local = idx & 65535;
    const float* src = (m == 0) ? Wq : (m == 1) ? Wk : (m == 2) ? Wv : Wo;
    Wb[idx] = (__bf16)src[local];
}

// ---------------- Kernel T: x[b,c,i,n] f32 -> xT[b,i,n,c] bf16 -----------------
__global__ void vn_transpose_x(const float* __restrict__ x, __bf16* __restrict__ xT) {
    __shared__ float t[32][33];
    const int b = blockIdx.z / 3, i = blockIdx.z % 3;
    const int n0 = blockIdx.x * 32, c0 = blockIdx.y * 32;
    const int tx = threadIdx.x, ty = threadIdx.y;
#pragma unroll
    for (int k = 0; k < 4; ++k) {
        const int c = c0 + ty + 8 * k;
        t[ty + 8 * k][tx] = x[((b * CC + c) * 3 + i) * NN + n0 + tx];
    }
    __syncthreads();
#pragma unroll
    for (int k = 0; k < 4; ++k) {
        const int n = n0 + ty + 8 * k;
        xT[((b * 3 + i) * NN + n) * CC + c0 + tx] = (__bf16)t[tx][ty + 8 * k];
    }
}

// ---------------- Kernel A: QKV projection, 2x4 register-blocked WMMA ----------
// wave 0 -> Q[b,h,n,96] (pre-scaled by 1/sqrt(HD)), wave 1 -> K[b,h,n,96],
// wave 2 -> Vt[b,h,96,n].  Each wave computes a 32(o) x 64(n) output block.
__global__ void vn_proj_qkv(const __bf16* __restrict__ xT,
                            const __bf16* __restrict__ Wb,   // [4][256][256] bf16
                            __bf16* __restrict__ Qb,
                            __bf16* __restrict__ Kb,
                            __bf16* __restrict__ Vt) {
    const int wave = threadIdx.x >> 5;
    const int lane = threadIdx.x & 31;
    const int l15 = lane & 15, hi = lane >> 4;
    const int b = blockIdx.z / 3, i = blockIdx.z % 3;
    const int o0 = blockIdx.y * 32, n0 = blockIdx.x * 64;

    const __bf16* W = Wb + (size_t)wave * CC * CC;           // Wq / Wk / Wv
    const __bf16* wrow = W + (size_t)(o0 + l15) * CC;
    const __bf16* xbase = xT + (size_t)(b * 3 + i) * NN * CC;

    v8f acc[2][4];
#pragma unroll
    for (int j = 0; j < 2; ++j)
#pragma unroll
        for (int k = 0; k < 4; ++k) acc[j][k] = (v8f){};

#pragma unroll
    for (int c0 = 0; c0 < CC; c0 += 32) {
        v16bf a0 = ldA(wrow + c0 + 8 * hi);
        v16bf a1 = ldA(wrow + (size_t)16 * CC + c0 + 8 * hi);
#pragma unroll
        for (int k = 0; k < 4; ++k) {
            v16bf bm = ldB(xbase + (size_t)(n0 + 16 * k + l15) * CC + c0 + 8 * hi);
            acc[0][k] = WMMA_BF16(a0, bm, acc[0][k]);
            acc[1][k] = WMMA_BF16(a1, bm, acc[1][k]);
        }
    }

    const float qscale = 0.17677669529663687f;  // 1/sqrt(32), folded into Q
#pragma unroll
    for (int j = 0; j < 2; ++j) {
        const int oj = o0 + 16 * j;
        const int h = oj >> 5;
        if (wave < 2) {
            const int cbase = (oj & 31) + 8 * hi;
#pragma unroll
            for (int k = 0; k < 4; ++k) {
                const int n = n0 + 16 * k + l15;
                __bf16* dst = (wave == 0 ? Qb : Kb) +
                              ((size_t)(b * HH + h) * NN + n) * K96 + i * HD + cbase;
                v8bf pk;
#pragma unroll
                for (int r = 0; r < 8; ++r)
                    pk[r] = (__bf16)(wave == 0 ? acc[j][k][r] * qscale : acc[j][k][r]);
                *(v8bf*)dst = pk;
            }
        } else {
#pragma unroll
            for (int k = 0; k < 4; ++k) {
                const int n = n0 + 16 * k + l15;
#pragma unroll
                for (int r = 0; r < 8; ++r) {
                    const int cp = (oj & 31) + 8 * hi + r;
                    Vt[((size_t)(b * HH + h) * K96 + i * HD + cp) * NN + n] =
                        (__bf16)acc[j][k][r];
                }
            }
        }
    }
}

// ---------------- Kernel B: flash attention ------------------------------------
// block = 4 waves x 16 query rows; key tiles of 32 staged in LDS; online softmax.
__global__ void vn_flash_attn(const __bf16* __restrict__ Qb,
                              const __bf16* __restrict__ Kb,
                              const __bf16* __restrict__ Vt,
                              __bf16* __restrict__ Ot) {
    __shared__ __bf16 kls[32 * KPAD];       // K tile: [row 0..31][96] padded
    __shared__ __bf16 vls[96 * VPAD];       // V tile: [c96 0..95][32] padded
    __shared__ __bf16 pbuf[4][16 * 32];     // per-wave P transpose scratch
    const int tid = threadIdx.x;
    const int w = tid >> 5;
    const int lane = tid & 31;
    const int l15 = lane & 15, hi = lane >> 4;
    const int h = blockIdx.y, b = blockIdx.z;
    const int q0 = blockIdx.x * 64 + w * 16;

    const __bf16* Qp = Qb + (size_t)(b * HH + h) * NN * K96;
    const __bf16* Kp = Kb + (size_t)(b * HH + h) * NN * K96;
    const __bf16* Vp = Vt + (size_t)(b * HH + h) * K96 * NN;

    // Q A-operands, 3 chunks of K=32 (Q already carries the softmax scale)
    v16bf qa[3];
#pragma unroll
    for (int t = 0; t < 3; ++t)
        qa[t] = ldA(Qp + (size_t)(q0 + l15) * K96 + 32 * t + 8 * hi);

    float mrow[8], lrow[8];
    v8f acc[6];
#pragma unroll
    for (int r = 0; r < 8; ++r) { mrow[r] = -1e30f; lrow[r] = 0.f; }
#pragma unroll
    for (int cc = 0; cc < 6; ++cc) acc[cc] = (v8f){};

    __bf16* pw = &pbuf[w][0];

    for (int k0 = 0; k0 < NN; k0 += 32) {
        // ---- cooperative staging of the K/V tile into LDS ----
        __syncthreads();   // previous iteration's readers are done
        if (tid < 96) {    // V rows: 96 x 32 halves (64 B each)
            const uint4* src = (const uint4*)(Vp + (size_t)tid * NN + k0);
            uint4* dst = (uint4*)(vls + tid * VPAD);
#pragma unroll
            for (int j = 0; j < 4; ++j) dst[j] = src[j];
        } else {           // K rows: 32 x 96 halves (192 B each)
            const int row = tid - 96;
            const uint4* src = (const uint4*)(Kp + (size_t)(k0 + row) * K96);
            uint4* dst = (uint4*)(kls + row * KPAD);
#pragma unroll
            for (int j = 0; j < 12; ++j) dst[j] = src[j];
        }
        __syncthreads();

        // ---- S = Q * K^T : two 16x16 tiles ----
        v8f s0 = {}, s1 = {};
#pragma unroll
        for (int t = 0; t < 3; ++t) {
            v16bf b0 = ldB(kls + (l15)*KPAD      + 32 * t + 8 * hi);
            v16bf b1 = ldB(kls + (16 + l15)*KPAD + 32 * t + 8 * hi);
            s0 = WMMA_BF16(qa[t], b0, s0);
            s1 = WMMA_BF16(qa[t], b1, s1);
        }
        // ---- online softmax per row (rows live in 16-lane groups) ----
#pragma unroll
        for (int r = 0; r < 8; ++r) {
            float a0 = s0[r], a1 = s1[r];
            float mx = fmaxf(a0, a1);
#pragma unroll
            for (int m = 8; m >= 1; m >>= 1) mx = fmaxf(mx, __shfl_xor(mx, m, 32));
            float mnew = fmaxf(mrow[r], mx);
            float corr = __expf(mrow[r] - mnew);
            mrow[r] = mnew;
            float p0 = __expf(a0 - mnew), p1 = __expf(a1 - mnew);
            float ps = p0 + p1;
#pragma unroll
            for (int m = 8; m >= 1; m >>= 1) ps += __shfl_xor(ps, m, 32);
            lrow[r] = lrow[r] * corr + ps;
#pragma unroll
            for (int cc = 0; cc < 6; ++cc) acc[cc][r] *= corr;
            pw[(r + 8 * hi) * 32 + l15]      = (__bf16)p0;
            pw[(r + 8 * hi) * 32 + 16 + l15] = (__bf16)p1;
        }
        // P back as A-operand (same-wave LDS is in-order)
        v16bf pa = ldA(pw + l15 * 32 + 8 * hi);
        // ---- O += P * V ----
#pragma unroll
        for (int cc = 0; cc < 6; ++cc) {
            v16bf vb = ldB(vls + (cc * 16 + l15) * VPAD + 8 * hi);
            acc[cc] = WMMA_BF16(pa, vb, acc[cc]);
        }
    }

    // normalize and scatter to Ot[b,i,n,256]
#pragma unroll
    for (int cc = 0; cc < 6; ++cc) {
        const int col = cc * 16 + l15;      // 0..95
        const int i = col >> 5, cp = col & 31;
#pragma unroll
        for (int r = 0; r < 8; ++r) {
            const int n = q0 + r + 8 * hi;
            float v = acc[cc][r] / lrow[r];
            Ot[((size_t)(b * 3 + i) * NN + n) * CC + h * HD + cp] = (__bf16)v;
        }
    }
}

// ---------------- Kernel C: out = Wo * O + x, 2x4 register-blocked -------------
__global__ void vn_proj_out(const __bf16* __restrict__ Ot,
                            const __bf16* __restrict__ Wob,  // bf16 Wo
                            const float* __restrict__ x,
                            float* __restrict__ out) {
    const int w = threadIdx.x >> 5;
    const int lane = threadIdx.x & 31;
    const int l15 = lane & 15, hi = lane >> 4;
    const int b = blockIdx.z / 3, i = blockIdx.z % 3;
    const int o0 = blockIdx.y * 32;
    const int n0 = (blockIdx.x * 4 + w) * 64;

    const __bf16* obase = Ot + (size_t)(b * 3 + i) * NN * CC;
    const __bf16* wrow = Wob + (size_t)(o0 + l15) * CC;

    v8f acc[2][4];
#pragma unroll
    for (int j = 0; j < 2; ++j)
#pragma unroll
        for (int k = 0; k < 4; ++k) acc[j][k] = (v8f){};

#pragma unroll
    for (int c0 = 0; c0 < CC; c0 += 32) {
        v16bf a0 = ldA(wrow + c0 + 8 * hi);
        v16bf a1 = ldA(wrow + (size_t)16 * CC + c0 + 8 * hi);
#pragma unroll
        for (int k = 0; k < 4; ++k) {
            v16bf bm = ldB(obase + (size_t)(n0 + 16 * k + l15) * CC + c0 + 8 * hi);
            acc[0][k] = WMMA_BF16(a0, bm, acc[0][k]);
            acc[1][k] = WMMA_BF16(a1, bm, acc[1][k]);
        }
    }

#pragma unroll
    for (int j = 0; j < 2; ++j)
#pragma unroll
        for (int k = 0; k < 4; ++k)
#pragma unroll
            for (int r = 0; r < 8; ++r) {
                const int o = o0 + 16 * j + r + 8 * hi;
                const size_t idx =
                    ((size_t)(b * CC + o) * 3 + i) * NN + n0 + 16 * k + l15;
                out[idx] = acc[j][k][r] + x[idx];
            }
}

extern "C" void kernel_launch(void* const* d_in, const int* in_sizes, int n_in,
                              void* d_out, int out_size, void* d_ws, size_t ws_size,
                              hipStream_t stream) {
    const float* x  = (const float*)d_in[0];
    const float* Wq = (const float*)d_in[1];
    const float* Wk = (const float*)d_in[2];
    const float* Wv = (const float*)d_in[3];
    const float* Wo = (const float*)d_in[4];
    float* out = (float*)d_out;

    const size_t S = (size_t)BB * 3 * NN * CC * sizeof(__bf16); // 12.58 MB each
    char* ws = (char*)d_ws;
    __bf16* xT = (__bf16*)(ws + 0 * S);
    __bf16* Qb = (__bf16*)(ws + 1 * S);
    __bf16* Kb = (__bf16*)(ws + 2 * S);
    __bf16* Vt = (__bf16*)(ws + 3 * S);
    __bf16* Ot = (__bf16*)(ws + 4 * S);
    __bf16* Wb = (__bf16*)(ws + 5 * S);     // 4 x 256 x 256 bf16 (512 KB)

    vn_cvt_w<<<dim3(4 * CC * CC / 256), 256, 0, stream>>>(Wq, Wk, Wv, Wo, Wb);
    vn_transpose_x<<<dim3(NN / 32, CC / 32, BB * 3), dim3(32, 8), 0, stream>>>(x, xT);
    vn_proj_qkv<<<dim3(NN / 64, CC / 32, BB * 3), 96, 0, stream>>>(xT, Wb, Qb, Kb, Vt);
    vn_flash_attn<<<dim3(NN / 64, HH, BB), 128, 0, stream>>>(Qb, Kb, Vt, Ot);
    vn_proj_out<<<dim3(NN / 256, CC / 32, BB * 3), 128, 0, stream>>>(
        Ot, Wb + 3 * (size_t)CC * CC, x, out);
}
